// LocalizedCrossAttentionBlock_18519898981035
// MI455X (gfx1250) — compile-verified, matrix operand order checked by reading, not run on hardware
//
#include <hip/hip_runtime.h>
#include <cstdint>

// ---------------------------------------------------------------------------
// MI455X (gfx1250) implementation of LocalizedCrossAttentionBlock.
// - All GEMMs + attention on v_wmma_f32_16x16x32_bf16 (wave32 WMMA)
// - Tile staging global->LDS via Tensor Data Mover (tensor_load_to_lds),
//   double-buffered, synchronized with s_wait_tensorcnt
// - Flash-attention streaming softmax (no N x N score materialization)
// ---------------------------------------------------------------------------

typedef __attribute__((ext_vector_type(16))) __bf16 v16bf;
typedef __attribute__((ext_vector_type(8)))  float  v8f;
typedef __attribute__((ext_vector_type(4))) unsigned int tdm_g0_t;
typedef __attribute__((ext_vector_type(8))) int          tdm_g1_t;
typedef __attribute__((ext_vector_type(4))) int          tdm_g2_t;

#define NTOK  2048
#define DIM   1024
#define NHEAD 16
#define HDIM  64

#define BM 128
#define BN 64
#define BK 64

#define BQ  128
#define BKV 64

__device__ __forceinline__ unsigned short f2bf(float f) {
  unsigned int u = __builtin_bit_cast(unsigned int, f);
  u += 0x7fffu + ((u >> 16) & 1u);                 // round-to-nearest-even
  return (unsigned short)(u >> 16);
}

__device__ __forceinline__ v8f wmma_bf16(v16bf a, v16bf b, v8f c) {
  return __builtin_amdgcn_wmma_f32_16x16x32_bf16(false, a, false, b, (short)0, c,
                                                 false, false);
}

// ---------------------------------------------------------------------------
// Tensor Data Mover: 2D bf16 tile (tile_d0 elems x tile_d1 rows) from global
// (row pitch = stride0 elems) into LDS at byte offset lds_off.
// D# per CDNA5 ISA 8.3/8.4: count=1, type=2 ("image"), data_size=1 (2 bytes).
// This toolchain exposes the 6-arg builtin:
//   (uint32x4 g0, int32x8 g1, int32x4 g2, int32x4 g3, int32x8, i32 cpol)
// ---------------------------------------------------------------------------
__device__ __forceinline__ void tdm_load_2d(unsigned long long gaddr,
                                            unsigned lds_off, unsigned tile_d0,
                                            unsigned tile_d1,
                                            unsigned long long stride0) {
  tdm_g0_t g0;
  g0[0] = 1u;                                          // count=1, user D#
  g0[1] = lds_off;                                     // lds_addr [63:32]
  g0[2] = (unsigned)(gaddr & 0xffffffffu);             // global_addr [95:64]
  g0[3] = (unsigned)((gaddr >> 32) & 0x01ffffffu) | (2u << 30);  // type=2
  tdm_g1_t g1;
  g1[0] = (int)(1u << 16);                             // data_size=2B, mask=0
  g1[1] = (int)((tile_d0 & 0xffffu) << 16);            // tensor_dim0 lo16
  g1[2] = (int)(((tile_d0 >> 16) & 0xffffu) |
                ((tile_d1 & 0xffffu) << 16));          // d0 hi | tensor_dim1 lo
  g1[3] = (int)(((tile_d1 >> 16) & 0xffffu) |
                ((tile_d0 & 0xffffu) << 16));          // d1 hi | tile_dim0
  g1[4] = (int)(tile_d1 & 0xffffu);                    // tile_dim1 (dim2=0)
  g1[5] = (int)(stride0 & 0xffffffffu);                // dim0_stride lo32
  g1[6] = (int)((stride0 >> 32) & 0xffffu);            // dim0_stride hi16
  g1[7] = 0;
  const tdm_g2_t gz4 = {0, 0, 0, 0};
  const tdm_g1_t gz8 = {0, 0, 0, 0, 0, 0, 0, 0};
  __builtin_amdgcn_tensor_load_to_lds(g0, g1, gz4, gz4, gz8, 0);
}

__device__ __forceinline__ void tdm_wait() {
  __builtin_amdgcn_s_wait_tensorcnt(0);
}

__device__ __forceinline__ unsigned lds_off_of(const void* p) {
  return (unsigned)(unsigned long long)p;              // low 32 bits = LDS addr
}

union FragU { v16bf v; unsigned int u[8]; unsigned short h[16]; };

// 16x32 bf16 fragment, rows = contiguous rows of a row-major LDS tile
// (ISA A-layout: lane = row, VGPR/half select K pairs). Feeding W rows as the
// B operand yields A * W^T.
__device__ __forceinline__ v16bf load_frag_rows(const unsigned short* lds,
                                                int row0, int k0, int ld) {
  const int lane = threadIdx.x & 31;
  const int m = lane & 15, hh = lane >> 4;
  FragU r;
  const unsigned short* base = lds + (row0 + m) * ld + k0;
#pragma unroll
  for (int v = 0; v < 8; ++v) {
    const int k = ((v & 4) ? 16 : 0) + hh * 8 + (v & 3) * 2;
    r.u[v] = *(const unsigned int*)(base + k);
  }
  return r.v;
}

// 32x16 B fragment from a row-major [K x N] LDS tile (true NN, for P·V).
__device__ __forceinline__ v16bf load_frag_cols(const unsigned short* lds,
                                                int k0, int n0, int ld) {
  const int lane = threadIdx.x & 31;
  const int n = n0 + (lane & 15), hh = lane >> 4;
  FragU r;
#pragma unroll
  for (int v = 0; v < 8; ++v) {
    const int k = k0 + ((v & 4) ? 16 : 0) + hh * 8 + (v & 3) * 2;
    r.h[2 * v]     = lds[k * ld + n];
    r.h[2 * v + 1] = lds[(k + 1) * ld + n];
  }
  return r.v;
}

// ---------------------------------------------------------------------------
__global__ void f32_to_bf16_kernel(const float* __restrict__ in,
                                   unsigned short* __restrict__ out, int n) {
  for (int i = blockIdx.x * blockDim.x + threadIdx.x; i < n;
       i += gridDim.x * blockDim.x)
    out[i] = f2bf(in[i]);
}

// ---------------------------------------------------------------------------
// GEMM: C[M,Nout] = A[M,K](bf16) * W[Nout,K](bf16)^T + bias (+resid) (silu?)
// 128x64 tile, K-step 64, TDM-staged double-buffered LDS.
// 8 waves: wave w owns C rows [w*16, w*16+16); each wave DMAs its own A slice,
// waves 0..3 DMA the 4 W slices.
// ---------------------------------------------------------------------------
__global__ __launch_bounds__(256) void gemm_nt(
    const unsigned short* __restrict__ A, const unsigned short* __restrict__ W,
    const float* __restrict__ bias, const float* __restrict__ resid,
    float* __restrict__ outF, unsigned short* __restrict__ outB,
    int M, int Nout, int K, int fuse_silu) {
  __shared__ unsigned short sA[2][BM * BK];  // 2 x 16 KB
  __shared__ unsigned short sW[2][BN * BK];  // 2 x  8 KB

  const int tid   = threadIdx.x;
  const int wave  = tid >> 5, lane = tid & 31;
  const int waveu = __builtin_amdgcn_readfirstlane(tid) >> 5;  // scalar wave id
  const int hh = lane >> 4, nl = lane & 15;
  const int row0 = blockIdx.x * BM;
  const int n0   = blockIdx.y * BN;

  v8f acc[4] = {};

  auto issue = [&](int buf, int k0) {
    tdm_load_2d(
        (unsigned long long)(const void*)(A + (size_t)(row0 + waveu * 16) * K + k0),
        lds_off_of(&sA[buf][waveu * 16 * BK]), BK, 16, (unsigned long long)K);
    if (waveu < 4)
      tdm_load_2d(
          (unsigned long long)(const void*)(W + (size_t)(n0 + waveu * 16) * K + k0),
          lds_off_of(&sW[buf][waveu * 16 * BK]), BK, 16, (unsigned long long)K);
  };

  issue(0, 0);
  tdm_wait();
  __syncthreads();

  const int T = K / BK;
  for (int t = 0; t < T; ++t) {
    const int cur = t & 1;
    if (t + 1 < T) issue(cur ^ 1, (t + 1) * BK);   // DMA next tile during WMMA

    const v16bf a0 = load_frag_rows(sA[cur], wave * 16, 0, BK);
    const v16bf a1 = load_frag_rows(sA[cur], wave * 16, 32, BK);
#pragma unroll
    for (int j = 0; j < 4; ++j) {
      acc[j] = wmma_bf16(a0, load_frag_rows(sW[cur], j * 16, 0, BK), acc[j]);
      acc[j] = wmma_bf16(a1, load_frag_rows(sW[cur], j * 16, 32, BK), acc[j]);
    }
    if (t + 1 < T) tdm_wait();                     // own DMA for next tile done
    __syncthreads();                               // publish / recycle buffers
  }

  // Epilogue: bias -> (silu) -> (+resid) -> f32/bf16 stores
#pragma unroll
  for (int j = 0; j < 4; ++j) {
    const int n = n0 + j * 16 + nl;
    const float bv = bias ? bias[n] : 0.f;
#pragma unroll
    for (int r = 0; r < 8; ++r) {
      const int m = row0 + wave * 16 + hh * 8 + r;
      float v = acc[j][r] + bv;
      if (fuse_silu) v = v / (1.f + __expf(-v));
      if (resid) v += resid[(size_t)m * Nout + n];
      if (outF) outF[(size_t)m * Nout + n] = v;
      if (outB) outB[(size_t)m * Nout + n] = f2bf(v);
    }
  }
}

// ---------------------------------------------------------------------------
// Flash attention over the block-diagonal ragged mask.
// Block = 128 queries x 1 head; 8 waves each own 16 query rows and sweep the
// 64-key tiles. K/V tiles TDM-staged, double-buffered. Scores and P·V via
// WMMA; online softmax in registers; P re-shaped through per-wave LDS.
// ---------------------------------------------------------------------------
__global__ __launch_bounds__(256) void flash_attn(
    const unsigned short* __restrict__ Q, const unsigned short* __restrict__ Kb,
    const unsigned short* __restrict__ V, unsigned short* __restrict__ O,
    const int* __restrict__ batch, const unsigned char* __restrict__ act,
    int N, int D) {
  __shared__ unsigned short sQ[BQ * HDIM];           // 16 KB
  __shared__ unsigned short sK[2][BKV * HDIM];       // 2 x 8 KB
  __shared__ unsigned short sV[2][BKV * HDIM];       // 2 x 8 KB
  __shared__ unsigned short sP[8 * 16 * BKV];        // 16 KB (per-wave P)

  const int tid   = threadIdx.x;
  const int wave  = tid >> 5, lane = tid & 31;
  const int waveu = __builtin_amdgcn_readfirstlane(tid) >> 5;
  const int hh = lane >> 4, nl = lane & 15;
  const int h  = blockIdx.y;
  const int q0 = blockIdx.x * BQ;

  auto issueKV = [&](int buf, int kbase) {           // 8 rows per wave each
    tdm_load_2d(
        (unsigned long long)(const void*)(Kb + (size_t)(kbase + waveu * 8) * D + h * HDIM),
        lds_off_of(&sK[buf][waveu * 8 * HDIM]), HDIM, 8, (unsigned long long)D);
    tdm_load_2d(
        (unsigned long long)(const void*)(V + (size_t)(kbase + waveu * 8) * D + h * HDIM),
        lds_off_of(&sV[buf][waveu * 8 * HDIM]), HDIM, 8, (unsigned long long)D);
  };

  // Q tile once (16 rows per wave) + first K/V tile
  tdm_load_2d(
      (unsigned long long)(const void*)(Q + (size_t)(q0 + waveu * 16) * D + h * HDIM),
      lds_off_of(&sQ[waveu * 16 * HDIM]), HDIM, 16, (unsigned long long)D);
  issueKV(0, 0);
  tdm_wait();
  __syncthreads();

  v16bf qf[2];
  qf[0] = load_frag_rows(sQ, wave * 16, 0, HDIM);
  qf[1] = load_frag_rows(sQ, wave * 16, 32, HDIM);

  int qb[8], qa[8];
#pragma unroll
  for (int r = 0; r < 8; ++r) {
    const int qr = q0 + wave * 16 + hh * 8 + r;
    qb[r] = batch[qr];
    qa[r] = act[qr];
  }

  float mrow[8], lrow[8];
  v8f o[4] = {};
#pragma unroll
  for (int r = 0; r < 8; ++r) { mrow[r] = -3.0e38f; lrow[r] = 0.f; }

  const int nT = N / BKV;
  for (int kt = 0; kt < nT; ++kt) {
    const int cur = kt & 1;
    const int kbase = kt * BKV;
    if (kt + 1 < nT) issueKV(cur ^ 1, (kt + 1) * BKV);

    // S = Q·K^T (16 q x 64 keys, K-dim 64)
    v8f s[4];
#pragma unroll
    for (int j = 0; j < 4; ++j) {
      v8f z = {};
      z = wmma_bf16(qf[0], load_frag_rows(sK[cur], j * 16, 0, HDIM), z);
      z = wmma_bf16(qf[1], load_frag_rows(sK[cur], j * 16, 32, HDIM), z);
      s[j] = z;
    }

    // scale + ragged mask (batch/act direct from L2), row-max
    float tmax[8];
#pragma unroll
    for (int r = 0; r < 8; ++r) tmax[r] = -3.0e38f;
#pragma unroll
    for (int j = 0; j < 4; ++j) {
      const int kcol = kbase + j * 16 + nl;
      const int kbv = batch[kcol];
      const int kav = act[kcol];
#pragma unroll
      for (int r = 0; r < 8; ++r) {
        float sv = s[j][r] * 0.125f;               // 1/sqrt(64)
        const bool valid = kav && qa[r] && (kbv == qb[r]);
        sv = valid ? sv : -1.0e9f;
        s[j][r] = sv;
        tmax[r] = fmaxf(tmax[r], sv);
      }
    }
#pragma unroll
    for (int off = 8; off >= 1; off >>= 1)
#pragma unroll
      for (int r = 0; r < 8; ++r)
        tmax[r] = fmaxf(tmax[r], __shfl_xor(tmax[r], off, 32));

    // online-softmax rescale
    float corr[8];
#pragma unroll
    for (int r = 0; r < 8; ++r) {
      const float mn = fmaxf(mrow[r], tmax[r]);
      corr[r] = __expf(mrow[r] - mn);
      mrow[r] = mn;
      lrow[r] *= corr[r];
    }
#pragma unroll
    for (int j = 0; j < 4; ++j)
#pragma unroll
      for (int r = 0; r < 8; ++r) o[j][r] *= corr[r];

    // P = exp(S - m) -> bf16 staged in this wave's LDS region (A-layout src)
    float rsum[8] = {0.f, 0.f, 0.f, 0.f, 0.f, 0.f, 0.f, 0.f};
    unsigned short* pbase = sP + wave * 16 * BKV;
#pragma unroll
    for (int j = 0; j < 4; ++j) {
#pragma unroll
      for (int r = 0; r < 8; ++r) {
        const float pv = __expf(s[j][r] - mrow[r]);
        rsum[r] += pv;
        pbase[(hh * 8 + r) * BKV + j * 16 + nl] = f2bf(pv);
      }
    }
#pragma unroll
    for (int off = 8; off >= 1; off >>= 1)
#pragma unroll
      for (int r = 0; r < 8; ++r) rsum[r] += __shfl_xor(rsum[r], off, 32);
#pragma unroll
    for (int r = 0; r < 8; ++r) lrow[r] += rsum[r];

    // O += P·V
#pragma unroll
    for (int ks = 0; ks < 2; ++ks) {
      const v16bf a = load_frag_rows(sP, wave * 16, ks * 32, BKV);
#pragma unroll
      for (int j = 0; j < 4; ++j) {
        const v16bf b = load_frag_cols(sV[cur], ks * 32, j * 16, HDIM);
        o[j] = wmma_bf16(a, b, o[j]);
      }
    }
    if (kt + 1 < nT) tdm_wait();
    __syncthreads();
  }

  // O /= l; write bf16
#pragma unroll
  for (int j = 0; j < 4; ++j) {
    const int c = h * HDIM + j * 16 + nl;
#pragma unroll
    for (int r = 0; r < 8; ++r) {
      const int qr = q0 + wave * 16 + hh * 8 + r;
      O[(size_t)qr * D + c] = f2bf(o[j][r] / lrow[r]);
    }
  }
}

// ---------------------------------------------------------------------------
// LayerNorm over D=1024 (one block per row), optional active-mask select.
// ---------------------------------------------------------------------------
__global__ __launch_bounds__(256) void layernorm_kernel(
    const float* __restrict__ x, const float* __restrict__ w,
    const float* __restrict__ b, float* __restrict__ outF,
    unsigned short* __restrict__ outB, const unsigned char* __restrict__ act,
    const float* __restrict__ orig, int Dd) {
  const int row = blockIdx.x;
  const int tid = threadIdx.x;
  const int wave = tid >> 5, lane = tid & 31;
  __shared__ float wsum[8], wsum2[8];

  float vals[4];
  float s = 0.f, s2 = 0.f;
#pragma unroll
  for (int i = 0; i < 4; ++i) {
    const float v = x[(size_t)row * Dd + tid + i * 256];
    vals[i] = v;
    s += v;
    s2 += v * v;
  }
#pragma unroll
  for (int off = 16; off >= 1; off >>= 1) {
    s += __shfl_xor(s, off, 32);
    s2 += __shfl_xor(s2, off, 32);
  }
  if (lane == 0) { wsum[wave] = s; wsum2[wave] = s2; }
  __syncthreads();
  float ts = 0.f, ts2 = 0.f;
#pragma unroll
  for (int i = 0; i < 8; ++i) { ts += wsum[i]; ts2 += wsum2[i]; }
  const float mean = ts / (float)Dd;
  float var = ts2 / (float)Dd - mean * mean;
  var = fmaxf(var, 0.f);
  const float rstd = rsqrtf(var + 1e-5f);
  const bool drop = (act != nullptr) && (act[row] == 0);
#pragma unroll
  for (int i = 0; i < 4; ++i) {
    const int c = tid + i * 256;
    float y = (vals[i] - mean) * rstd * w[c] + b[c];
    if (drop) y = orig[(size_t)row * Dd + c];
    if (outF) outF[(size_t)row * Dd + c] = y;
    if (outB) outB[(size_t)row * Dd + c] = f2bf(y);
  }
}

// ---------------------------------------------------------------------------
// Host launch
// ---------------------------------------------------------------------------
extern "C" void kernel_launch(void* const* d_in, const int* in_sizes, int n_in,
                              void* d_out, int out_size, void* d_ws,
                              size_t ws_size, hipStream_t stream) {
  (void)in_sizes; (void)n_in; (void)out_size; (void)ws_size;
  const int N = NTOK, D = DIM;
  const size_t ND = (size_t)N * D;

  const float* x_struct = (const float*)d_in[0];
  const float* x_esm    = (const float*)d_in[1];
  const int*   batch    = (const int*)d_in[2];
  const unsigned char* act = (const unsigned char*)d_in[3];

  const float* wf[2][4];
  const float* bf[2][8];
  for (int s = 0; s < 2; ++s) {
    const int o = 4 + s * 12;
    wf[s][0] = (const float*)d_in[o + 0];   // in_w  (3D x D)
    bf[s][0] = (const float*)d_in[o + 1];   // in_b
    wf[s][1] = (const float*)d_in[o + 2];   // out_w
    bf[s][1] = (const float*)d_in[o + 3];   // out_b
    bf[s][2] = (const float*)d_in[o + 4];   // ln1_w
    bf[s][3] = (const float*)d_in[o + 5];   // ln1_b
    wf[s][2] = (const float*)d_in[o + 6];   // ff1_w
    bf[s][4] = (const float*)d_in[o + 7];   // ff1_b
    wf[s][3] = (const float*)d_in[o + 8];   // ff2_w
    bf[s][5] = (const float*)d_in[o + 9];   // ff2_b
    bf[s][6] = (const float*)d_in[o + 10];  // ln2_w
    bf[s][7] = (const float*)d_in[o + 11];  // ln2_b
  }
  float* outF = (float*)d_out;

  char* p = (char*)d_ws;
  auto alloc = [&](size_t bytes) -> void* {
    void* r = (void*)p;
    p += (bytes + 255) & ~(size_t)255;
    return r;
  };
  unsigned short* xs_bf = (unsigned short*)alloc(ND * 2);
  unsigned short* xe_bf = (unsigned short*)alloc(ND * 2);
  unsigned short* wbf[2][4];
  for (int s = 0; s < 2; ++s) {
    wbf[s][0] = (unsigned short*)alloc((size_t)3 * D * D * 2);
    wbf[s][1] = (unsigned short*)alloc((size_t)D * D * 2);
    wbf[s][2] = (unsigned short*)alloc((size_t)D * D * 2);
    wbf[s][3] = (unsigned short*)alloc((size_t)D * D * 2);
  }
  unsigned short* q_bf    = (unsigned short*)alloc(ND * 2);
  unsigned short* k_bf    = (unsigned short*)alloc(ND * 2);
  unsigned short* v_bf    = (unsigned short*)alloc(ND * 2);
  unsigned short* attn_bf = (unsigned short*)alloc(ND * 2);
  unsigned short* xln_bf  = (unsigned short*)alloc(ND * 2);
  unsigned short* h_bf    = (unsigned short*)alloc(ND * 2);
  unsigned short* upds_bf = (unsigned short*)alloc(ND * 2);
  float* res1 = (float*)alloc(ND * 4);
  float* xlnf = (float*)alloc(ND * 4);
  float* res2 = (float*)alloc(ND * 4);

  auto cvt = [&](const float* src, unsigned short* dst, size_t n) {
    f32_to_bf16_kernel<<<dim3(512), dim3(256), 0, stream>>>(src, dst, (int)n);
  };
  cvt(x_struct, xs_bf, ND);
  cvt(x_esm, xe_bf, ND);
  for (int s = 0; s < 2; ++s) {
    cvt(wf[s][0], wbf[s][0], (size_t)3 * D * D);
    cvt(wf[s][1], wbf[s][1], (size_t)D * D);
    cvt(wf[s][2], wbf[s][2], (size_t)D * D);
    cvt(wf[s][3], wbf[s][3], (size_t)D * D);
  }

  const dim3 gG(N / BM, D / BN);   // 16 x 16
  const dim3 gA(N / BQ, NHEAD);    // 16 x 16

  auto phase = [&](const unsigned short* qx, const unsigned short* kvx,
                   const float* x_orig, int s, float* oF, unsigned short* oB) {
    gemm_nt<<<gG, 256, 0, stream>>>(qx,  wbf[s][0],                     bf[s][0],         nullptr, nullptr, q_bf, N, D, D, 0);
    gemm_nt<<<gG, 256, 0, stream>>>(kvx, wbf[s][0] + (size_t)D * D,     bf[s][0] + D,     nullptr, nullptr, k_bf, N, D, D, 0);
    gemm_nt<<<gG, 256, 0, stream>>>(kvx, wbf[s][0] + (size_t)2 * D * D, bf[s][0] + 2 * D, nullptr, nullptr, v_bf, N, D, D, 0);
    flash_attn<<<gA, 256, 0, stream>>>(q_bf, k_bf, v_bf, attn_bf, batch, act, N, D);
    gemm_nt<<<gG, 256, 0, stream>>>(attn_bf, wbf[s][1], bf[s][1], x_orig, res1, nullptr, N, D, D, 0);
    layernorm_kernel<<<N, 256, 0, stream>>>(res1, bf[s][2], bf[s][3], xlnf, xln_bf, nullptr, nullptr, D);
    gemm_nt<<<gG, 256, 0, stream>>>(xln_bf, wbf[s][2], bf[s][4], nullptr, nullptr, h_bf, N, D, D, 1);
    gemm_nt<<<gG, 256, 0, stream>>>(h_bf,   wbf[s][3], bf[s][5], xlnf, res2, nullptr, N, D, D, 0);
    layernorm_kernel<<<N, 256, 0, stream>>>(res2, bf[s][6], bf[s][7], oF, oB, act, x_orig, D);
  };

  phase(xs_bf, xe_bf, x_struct, 0, outF, upds_bf);        // struct <- esm
  phase(xe_bf, upds_bf, x_esm, 1, outF + ND, nullptr);    // esm <- updated struct
}
